// NestedGIN_27522150432785
// MI455X (gfx1250) — matrix operand order repaired and checked by reading, not compiled, not run on hardware
//
#include <hip/hip_runtime.h>
#include <hip/hip_bf16.h>
#include <stdint.h>

typedef __attribute__((ext_vector_type(16))) __bf16 v16bf;
typedef __attribute__((ext_vector_type(8)))  __bf16 v8bf;
typedef __attribute__((ext_vector_type(8)))  float  v8f;

#define N_NODES 100000
#define E_EDGES 1600000
#define D_DIM   256
#define L_LAYERS 3
#define S_SUB   5000
#define G_GRAPHS 64
#define OUT_DIM 64
#define CAT (L_LAYERS * D_DIM)   // 768

#define CEILDIV(a, b) (((a) + (b) - 1) / (b))

// ---------------------------------------------------------------------------
// utility kernels
// ---------------------------------------------------------------------------
__global__ void zero_f32(float* __restrict__ p, int n) {
    int i = blockIdx.x * blockDim.x + threadIdx.x;
    if (i < n) p[i] = 0.0f;
}

// wt layout: [l][m][n][k] bf16, m=0 -> Ws1[l], m=1 -> Ws2[l]; wt = W[l][k][n] transposed
__global__ void convert_weights(const float* __restrict__ Ws1,
                                const float* __restrict__ Ws2,
                                __bf16* __restrict__ wt) {
    int i = blockIdx.x * blockDim.x + threadIdx.x;
    if (i >= L_LAYERS * 2 * D_DIM * D_DIM) return;
    int k = i & 255;
    int n = (i >> 8) & 255;
    int m = (i >> 16) & 1;
    int l = i >> 17;
    const float* W = (m == 0 ? Ws1 : Ws2) + l * D_DIM * D_DIM;
    wt[i] = (__bf16)W[k * D_DIM + n];
}

__global__ void mark_mask(const int* __restrict__ ei, float* __restrict__ mask, int twoE) {
    int i = blockIdx.x * blockDim.x + threadIdx.x;
    if (i < twoE) mask[ei[i]] = 1.0f;
}

__global__ void count_nodes(const float* __restrict__ mask, const int* __restrict__ n2s,
                            float* __restrict__ cnt) {
    int i = blockIdx.x * blockDim.x + threadIdx.x;
    if (i < N_NODES && mask[i] != 0.0f) atomicAdd(&cnt[n2s[i]], 1.0f);
}

// y = (1 + eps[l]) * h   (vectorized float4)
__global__ void init_y4(const float* __restrict__ h, const float* __restrict__ eps, int l,
                        float* __restrict__ y) {
    int i = blockIdx.x * blockDim.x + threadIdx.x;
    if (i >= N_NODES * D_DIM / 4) return;
    float s = 1.0f + eps[l];
    float4 v = ((const float4*)h)[i];
    v.x *= s; v.y *= s; v.z *= s; v.w *= s;
    ((float4*)y)[i] = v;
}

// y[dst] += h[src]  over all edges; 4 floats per thread
__global__ void scatter_edges(const int* __restrict__ ei, const float* __restrict__ h,
                              float* __restrict__ y) {
    int idx = blockIdx.x * blockDim.x + threadIdx.x;
    if (idx >= E_EDGES * (D_DIM / 4)) return;
    int e = idx >> 6;            // D/4 = 64 chunks per edge
    int q = (idx & 63) << 2;
    int src = ei[e];
    int dst = ei[E_EDGES + e];
    const float4 v = *(const float4*)(h + src * D_DIM + q);
    float* o = y + dst * D_DIM + q;
    __builtin_prefetch(o, 1, 0);          // global_prefetch_b8 hint at the RMW target
    atomicAdd(o + 0, v.x);
    atomicAdd(o + 1, v.y);
    atomicAdd(o + 2, v.z);
    atomicAdd(o + 3, v.w);
}

// ---------------------------------------------------------------------------
// WMMA GEMM:  Out[M x 256] = act(A[M x 256] @ W[256 x 256] + bias)
//   Wt is pre-transposed bf16 [n][k] so B fragments are contiguous 32B loads.
//   BLOCK: 128 rows x 64 cols, 256 threads (8 waves), each wave: 16 rows x 4 n-tiles.
//   A_BF16=false : A is fp32, staged through LDS with f32->bf16 conversion.
//   A_BF16=true  : A is bf16; double-buffered GLOBAL_LOAD_ASYNC_TO_LDS_B128 pipeline
//                  (ASYNCcnt-tracked DMA overlapped with WMMA).
// ---------------------------------------------------------------------------
template <bool A_BF16, bool RELU, bool OUT_BF16>
__global__ __launch_bounds__(256) void gemm_wmma(const void* __restrict__ Aptr,
                                                 const __bf16* __restrict__ Wt,
                                                 const float* __restrict__ bias,
                                                 void* __restrict__ Out, int Mrows) {
    const int m0    = blockIdx.x * 128;
    const int n0    = blockIdx.y * 64;
    const int tid   = threadIdx.x;
    const int lane  = tid & 31;
    const int wave  = tid >> 5;
    const int wrow0 = wave * 16;

    // 2 x (128 rows x 32 K-cols + pad); buffer 0 doubles as the staging buffer
    // for the fp32 path. Row stride 40 keeps every 8-element group 16B aligned.
    __shared__ __bf16 As[2][128][40];

    v8f acc[4];
#pragma unroll
    for (int t = 0; t < 4; ++t)
#pragma unroll
        for (int j = 0; j < 8; ++j) acc[t][j] = 0.0f;

    const int r  = lane & 15;
    const int c0 = (lane >> 4) * 16;               // contiguous K split (matches B)
    const bool rowActive = (m0 + wrow0) < Mrows;   // M multiple of 16 -> wave-uniform

    // Async DMA of one 128x32 bf16 tile (8KB) = 512 x 16B chunks, 2 per thread.
    auto issue_async = [&](int ks, int buf) {
        const __bf16* A = (const __bf16*)Aptr;
#pragma unroll
        for (int j = 0; j < 2; ++j) {
            int c  = tid + j * 256;
            int rr = c >> 2;          // tile row 0..127
            int p  = c & 3;           // 8-element (16B) part within the 32-col slice
            int gr = m0 + rr;
            if (gr > Mrows - 1) gr = Mrows - 1;   // clamped tail (results discarded)
            const __bf16* gp = A + gr * D_DIM + ks * 32 + p * 8;
            unsigned lds = (unsigned)(uintptr_t)(&As[buf][rr][p * 8]);
            asm volatile("global_load_async_to_lds_b128 %0, %1, off"
                         :: "v"(lds), "v"((unsigned long long)(uintptr_t)gp)
                         : "memory");
        }
    };

    if constexpr (A_BF16) issue_async(0, 0);

    for (int ks = 0; ks < 8; ++ks) {               // K = 256 in steps of 32
        v16bf a;
        if constexpr (!A_BF16) {
            // stage 128x32 fp32 -> bf16 LDS; each thread: one row-half (16 floats)
            const float* A = (const float*)Aptr;
            int sr = tid >> 1;
            int sc = (tid & 1) * 16;
            int grow = m0 + sr;
            v8bf p0, p1;
            if (grow < Mrows) {
                const float4* base = (const float4*)(A + grow * D_DIM + ks * 32 + sc);
                float4 f0 = base[0], f1 = base[1], f2 = base[2], f3 = base[3];
                p0[0] = (__bf16)f0.x; p0[1] = (__bf16)f0.y; p0[2] = (__bf16)f0.z; p0[3] = (__bf16)f0.w;
                p0[4] = (__bf16)f1.x; p0[5] = (__bf16)f1.y; p0[6] = (__bf16)f1.z; p0[7] = (__bf16)f1.w;
                p1[0] = (__bf16)f2.x; p1[1] = (__bf16)f2.y; p1[2] = (__bf16)f2.z; p1[3] = (__bf16)f2.w;
                p1[4] = (__bf16)f3.x; p1[5] = (__bf16)f3.y; p1[6] = (__bf16)f3.z; p1[7] = (__bf16)f3.w;
            } else {
#pragma unroll
                for (int j = 0; j < 8; ++j) { p0[j] = (__bf16)0.0f; p1[j] = (__bf16)0.0f; }
            }
            *(v8bf*)&As[0][sr][sc]     = p0;
            *(v8bf*)&As[0][sr][sc + 8] = p1;
            __syncthreads();
            v8bf alo = *(const v8bf*)&As[0][wrow0 + r][c0];        // K = c0 .. c0+7
            v8bf ahi = *(const v8bf*)&As[0][wrow0 + r][c0 + 8];    // K = c0+8 .. c0+15
#pragma unroll
            for (int j = 0; j < 8; ++j) { a[j] = alo[j]; a[8 + j] = ahi[j]; }
        } else {
            // pipeline: kick off next stage, then drain current stage
            if (ks < 7) {
                issue_async(ks + 1, (ks + 1) & 1);
                asm volatile("s_wait_asynccnt 2" ::: "memory");  // next stage in flight
            } else {
                asm volatile("s_wait_asynccnt 0" ::: "memory");
            }
            __syncthreads();
            v8bf alo = *(const v8bf*)&As[ks & 1][wrow0 + r][c0];
            v8bf ahi = *(const v8bf*)&As[ks & 1][wrow0 + r][c0 + 8];
#pragma unroll
            for (int j = 0; j < 8; ++j) { a[j] = alo[j]; a[8 + j] = ahi[j]; }
        }

        // All waves execute WMMA (EXEC all-1s); tail waves' results are never stored.
#pragma unroll
        for (int nt = 0; nt < 4; ++nt) {
            const __bf16* bp =
                Wt + (n0 + nt * 16 + (lane & 15)) * D_DIM + ks * 32 + (lane >> 4) * 16;
            v16bf b = *(const v16bf*)bp;   // 32B contiguous (L2-resident weights)
            acc[nt] = __builtin_amdgcn_wmma_f32_16x16x32_bf16(
                false, a, false, b, (short)0, acc[nt], false, false);
        }
        __syncthreads();   // all reads of this buffer done before it is overwritten
    }

    if (!rowActive) return;
#pragma unroll
    for (int nt = 0; nt < 4; ++nt) {
        int col  = n0 + nt * 16 + (lane & 15);
        float bv = bias[col];
#pragma unroll
        for (int rr = 0; rr < 8; ++rr) {
            int row = m0 + wrow0 + rr + (lane >> 4) * 8;   // ISA f32 C/D layout
            float v = acc[nt][rr] + bv;
            if (RELU) v = fmaxf(v, 0.0f);
            if (OUT_BF16)
                ((__bf16*)Out)[row * D_DIM + col] = (__bf16)v;
            else
                ((float*)Out)[row * D_DIM + col] = v;
        }
    }
}

// ---------------------------------------------------------------------------
// pooling + readout
// ---------------------------------------------------------------------------
__global__ void pool_accum(const float* __restrict__ h, const float* __restrict__ mask,
                           const int* __restrict__ n2s, float* __restrict__ sums, int l) {
    int idx = blockIdx.x * blockDim.x + threadIdx.x;
    if (idx >= N_NODES * (D_DIM / 4)) return;
    int i = idx >> 6;
    int q = (idx & 63) << 2;
    if (mask[i] == 0.0f) return;
    int s = n2s[i];
    const float4 v = *(const float4*)(h + i * D_DIM + q);
    float* o = sums + s * CAT + l * D_DIM + q;
    atomicAdd(o + 0, v.x);
    atomicAdd(o + 1, v.y);
    atomicAdd(o + 2, v.z);
    atomicAdd(o + 3, v.w);
}

__global__ void graph_pool(const float* __restrict__ sums, const float* __restrict__ cnt,
                           const int* __restrict__ s2g, float* __restrict__ g) {
    int idx = blockIdx.x * blockDim.x + threadIdx.x;
    if (idx >= S_SUB * CAT) return;
    int s = idx / CAT;
    int f = idx - s * CAT;
    float c = fmaxf(cnt[s], 1.0f);
    atomicAdd(&g[s2g[s] * CAT + f], sums[idx] / c);
}

__global__ void fin1(const float* __restrict__ g, const float* __restrict__ w,
                     const float* __restrict__ b, float* __restrict__ gh) {
    int idx = blockIdx.x * blockDim.x + threadIdx.x;
    if (idx >= G_GRAPHS * D_DIM) return;
    int gi = idx >> 8, j = idx & 255;
    float acc = b[j];
    for (int k = 0; k < CAT; ++k) acc += g[gi * CAT + k] * w[k * D_DIM + j];
    gh[idx] = fmaxf(acc, 0.0f);
}

__global__ void fin2(const float* __restrict__ gh, const float* __restrict__ w,
                     const float* __restrict__ b, float* __restrict__ out) {
    int idx = blockIdx.x * blockDim.x + threadIdx.x;
    if (idx >= G_GRAPHS * OUT_DIM) return;
    int gi = idx >> 6, o = idx & 63;
    float acc = b[o];
    for (int k = 0; k < D_DIM; ++k) acc += gh[gi * D_DIM + k] * w[k * OUT_DIM + o];
    out[idx] = acc;
}

// ---------------------------------------------------------------------------
extern "C" void kernel_launch(void* const* d_in, const int* in_sizes, int n_in,
                              void* d_out, int out_size, void* d_ws, size_t ws_size,
                              hipStream_t stream) {
    (void)in_sizes; (void)n_in; (void)out_size; (void)ws_size;

    const float* x      = (const float*)d_in[0];
    const int*   ei     = (const int*)d_in[1];
    const int*   n2s    = (const int*)d_in[2];
    const int*   s2g    = (const int*)d_in[3];
    const float* Ws1    = (const float*)d_in[4];
    const float* bs1    = (const float*)d_in[5];
    const float* Ws2    = (const float*)d_in[6];
    const float* bs2    = (const float*)d_in[7];
    const float* eps    = (const float*)d_in[8];
    const float* lin1_w = (const float*)d_in[9];
    const float* lin1_b = (const float*)d_in[10];
    const float* lin2_w = (const float*)d_in[11];
    const float* lin2_b = (const float*)d_in[12];

    char* ws = (char*)d_ws;
    size_t off = 0;
    auto alloc = [&](size_t bytes) -> void* {
        void* p = ws + off;
        off = (off + bytes + 255) & ~(size_t)255;
        return p;
    };
    float*  y    = (float*)alloc(sizeof(float) * N_NODES * D_DIM);
    float*  hbuf = (float*)alloc(sizeof(float) * N_NODES * D_DIM);
    __bf16* z    = (__bf16*)alloc(sizeof(__bf16) * N_NODES * D_DIM);
    __bf16* wt   = (__bf16*)alloc(sizeof(__bf16) * L_LAYERS * 2 * D_DIM * D_DIM);
    float*  sums = (float*)alloc(sizeof(float) * S_SUB * CAT);
    float*  cnt  = (float*)alloc(sizeof(float) * S_SUB);
    float*  gbuf = (float*)alloc(sizeof(float) * G_GRAPHS * CAT);
    float*  gh   = (float*)alloc(sizeof(float) * G_GRAPHS * D_DIM);
    float*  mask = (float*)alloc(sizeof(float) * N_NODES);

    const int T = 256;

    // zero accumulators / mask
    zero_f32<<<CEILDIV(S_SUB * CAT, T), T, 0, stream>>>(sums, S_SUB * CAT);
    zero_f32<<<CEILDIV(S_SUB, T), T, 0, stream>>>(cnt, S_SUB);
    zero_f32<<<CEILDIV(G_GRAPHS * CAT, T), T, 0, stream>>>(gbuf, G_GRAPHS * CAT);
    zero_f32<<<CEILDIV(N_NODES, T), T, 0, stream>>>(mask, N_NODES);

    // bf16 transposed weights (L2-resident for all GEMMs)
    convert_weights<<<CEILDIV(L_LAYERS * 2 * D_DIM * D_DIM, T), T, 0, stream>>>(Ws1, Ws2, wt);

    // edge mask + per-subgraph masked counts
    mark_mask<<<CEILDIV(2 * E_EDGES, T), T, 0, stream>>>(ei, mask, 2 * E_EDGES);
    count_nodes<<<CEILDIV(N_NODES, T), T, 0, stream>>>(mask, n2s, cnt);

    const dim3 ggrid(CEILDIV(N_NODES, 128), D_DIM / 64);   // 782 x 4

    for (int l = 0; l < L_LAYERS; ++l) {
        const float* hcur = (l == 0) ? x : hbuf;
        // y = (1+eps)*h ; then y[dst] += h[src] over edges (fused GIN pre-MLP)
        init_y4<<<CEILDIV(N_NODES * D_DIM / 4, T), T, 0, stream>>>(hcur, eps, l, y);
        scatter_edges<<<CEILDIV(E_EDGES * (D_DIM / 4), T), T, 0, stream>>>(ei, hcur, y);
        // z = relu(y @ W1 + b1)  (bf16 out, fp32->bf16 staged through LDS)
        gemm_wmma<false, true, true><<<ggrid, T, 0, stream>>>(
            y, wt + (l * 2 + 0) * D_DIM * D_DIM, bs1 + l * D_DIM, z, N_NODES);
        // h = z @ W2 + b2  (fp32 out, async-DMA double-buffered A tiles)
        gemm_wmma<true, false, false><<<ggrid, T, 0, stream>>>(
            z, wt + (l * 2 + 1) * D_DIM * D_DIM, bs2 + l * D_DIM, hbuf, N_NODES);
        // masked per-subgraph sum of this layer's features
        pool_accum<<<CEILDIV(N_NODES * (D_DIM / 4), T), T, 0, stream>>>(hbuf, mask, n2s, sums, l);
    }

    graph_pool<<<CEILDIV(S_SUB * CAT, T), T, 0, stream>>>(sums, cnt, s2g, gbuf);
    fin1<<<CEILDIV(G_GRAPHS * D_DIM, T), T, 0, stream>>>(gbuf, lin1_w, lin1_b, gh);
    fin2<<<CEILDIV(G_GRAPHS * OUT_DIM, T), T, 0, stream>>>(gh, lin2_w, lin2_b, (float*)d_out);
}
